// TaskSpecificExperts_51926154609119
// MI455X (gfx1250) — compile-verified, matrix operand order checked by reading, not run on hardware
//
#include <hip/hip_runtime.h>
#include <stdint.h>

// ---- problem constants (match reference) -----------------------------------
#define DIMN   512      // model dim (K and O)
#define NEXP   16       // experts
#define NTOK   8192     // tokens

typedef _Float16 v8h  __attribute__((ext_vector_type(8)));
typedef _Float16 v16h __attribute__((ext_vector_type(16)));
typedef float    v8f  __attribute__((ext_vector_type(8)));

// Try the gfx1250 async global->LDS path (ASYNCcnt-tracked, no VGPR staging).
// Flip to 0 if the assembler rejects the mnemonic.
#define TRY_ASYNC_G2L 1

// Dynamic LDS layout:
//   A panel : [2 planes][32 rows][A_PITCH halves]   (hi, lo of x rows)
//   Out buf : [8 waves][32 rows][SOUT_PITCH floats] (store repack)
#define A_PITCH       520                       // 512 + 8 pad -> bank-conflict-free
#define A_PLANE       (32 * A_PITCH)            // halves per plane
#define SMEM_A_BYTES  (2 * A_PLANE * 2)         // 66,560 B
#define SOUT_PITCH    68                        // 64 + 4 pad floats (272 B, 16B-mult)
#define SMEM_OUT_BYTES (8 * 32 * SOUT_PITCH * 4)// 69,632 B
#define SMEM_BYTES    (SMEM_A_BYTES + SMEM_OUT_BYTES) // 136,192 B (<320 KB WGP LDS)

// ---------------------------------------------------------------------------
// Pass 1: split fp32 -> (f16 hi, f16 lo) planes.  lo = f16(x - f32(hi)).
// ---------------------------------------------------------------------------
__global__ void split_f32_to_f16hl(const float* __restrict__ src,
                                   _Float16* __restrict__ hi,
                                   _Float16* __restrict__ lo,
                                   int n) {
    int i = (blockIdx.x * blockDim.x + threadIdx.x) * 4;
    if (i >= n) return;
    float4 v = *(const float4*)(src + i);
    _Float16 h0 = (_Float16)v.x, h1 = (_Float16)v.y;
    _Float16 h2 = (_Float16)v.z, h3 = (_Float16)v.w;
    hi[i + 0] = h0; hi[i + 1] = h1; hi[i + 2] = h2; hi[i + 3] = h3;
    lo[i + 0] = (_Float16)(v.x - (float)h0);
    lo[i + 1] = (_Float16)(v.y - (float)h1);
    lo[i + 2] = (_Float16)(v.z - (float)h2);
    lo[i + 3] = (_Float16)(v.w - (float)h3);
}

// ---------------------------------------------------------------------------
// Async 16-byte global -> LDS copy (gfx1250 VGLOBAL async op, ASYNCcnt).
// VDST = per-lane LDS byte address, VADDR = 64-bit global address.
// ---------------------------------------------------------------------------
__device__ __forceinline__ void async_g2l_b128(void* lds_dst, const void* gsrc) {
#if TRY_ASYNC_G2L
    uint32_t l = (uint32_t)(uintptr_t)lds_dst;        // low 32 bits = LDS address
    uint64_t g = (uint64_t)(uintptr_t)gsrc;
    asm volatile("global_load_async_to_lds_b128 %0, %1, off"
                 :: "v"(l), "v"(g) : "memory");
#else
    *(v8h*)lds_dst = *(const v8h*)gsrc;               // load_b128 + ds_store_b128
#endif
}

__device__ __forceinline__ void wait_async_g2l() {
#if TRY_ASYNC_G2L
    asm volatile("s_wait_asynccnt 0x0" ::: "memory");
#endif
}

// ---------------------------------------------------------------------------
// Load one 16x32 f16 tile (K contiguous) into the WMMA operand layout:
//   lane<16 : row = lane,    K = 0..7  / 16..23
//   lane>=16: row = lane-16, K = 8..15 / 24..31
// Works for both global and LDS pointers (addrspace inferred after inlining).
// ---------------------------------------------------------------------------
__device__ __forceinline__ v16h load_tile16(const _Float16* __restrict__ base,
                                            int ld, int lane) {
    const int r    = lane & 15;
    const int koff = (lane < 16) ? 0 : 8;
    const _Float16* p = base + (size_t)r * ld + koff;
    v8h c0 = *(const v8h*)(p);
    v8h c1 = *(const v8h*)(p + 16);
    v16h t;
#pragma unroll
    for (int i = 0; i < 8; ++i) { t[i] = c0[i]; t[i + 8] = c1[i]; }
    return t;
}

#define WMMA_F16(A, B, C)                                                     \
    __builtin_amdgcn_wmma_f32_16x16x32_f16(false, (A), false, (B),            \
                                           (short)0, (C), false, false)

// ---------------------------------------------------------------------------
// Pass 2: block = 32 token rows x all 512 cols of one expert (8 waves x 64).
//   x hi/lo panel staged once in LDS (async), B streamed from L2,
//   acc += Ah*Bh + Al*Bh + Ah*Bl on v_wmma_f32_16x16x32_f16.
//   Epilogue repacks through LDS for 256B-contiguous global_store_b128.
// ---------------------------------------------------------------------------
__global__ __launch_bounds__(256)
void expert_gemm_wmma(const _Float16* __restrict__ xh,
                      const _Float16* __restrict__ xl,
                      const _Float16* __restrict__ wh,
                      const _Float16* __restrict__ wl,
                      const float* __restrict__ bias,
                      float* __restrict__ out) {
    extern __shared__ char smem[];
    _Float16* sA   = (_Float16*)smem;                  // [2][32][A_PITCH]
    float*    sOut = (float*)(smem + SMEM_A_BYTES);    // [8][32][SOUT_PITCH]

    const int tid  = threadIdx.x;
    const int lane = tid & 31;
    const int wid  = tid >> 5;          // 0..7 -> 64-col group
    const int mt   = blockIdx.x & 255;  // token tile
    const int e    = blockIdx.x >> 8;   // expert

    const int m0 = mt * 32;
    const int c0 = wid * 64;

    // ---- stage x hi/lo panel (32 x 512, both planes) into LDS -------------
    {
        const _Float16* srcs[2] = { xh + (size_t)m0 * DIMN, xl + (size_t)m0 * DIMN };
#pragma unroll
        for (int p = 0; p < 2; ++p) {
            const _Float16* src = srcs[p];
            _Float16* dst = sA + p * A_PLANE;
#pragma unroll
            for (int it = 0; it < 8; ++it) {
                int idx = it * 256 + tid;          // 2048 16B-chunks per plane
                int row = idx >> 6;                // 64 chunks per row
                int k8  = (idx & 63) << 3;
                async_g2l_b128(dst + row * A_PITCH + k8,
                               src + (size_t)row * DIMN + k8);
            }
        }
        wait_async_g2l();
        __syncthreads();
    }

    const _Float16* sAh = sA;
    const _Float16* sAl = sA + A_PLANE;
    const _Float16* Bh  = wh + ((size_t)e * DIMN + c0) * DIMN;
    const _Float16* Bl  = wl + ((size_t)e * DIMN + c0) * DIMN;

    v8f zero;
#pragma unroll
    for (int i = 0; i < 8; ++i) zero[i] = 0.0f;
    v8f acc[2][4];
#pragma unroll
    for (int r = 0; r < 2; ++r)
#pragma unroll
        for (int j = 0; j < 4; ++j) acc[r][j] = zero;

    for (int kc = 0; kc < DIMN; kc += 32) {
        v16h a0h = load_tile16(sAh + kc,               A_PITCH, lane);
        v16h a1h = load_tile16(sAh + 16 * A_PITCH + kc, A_PITCH, lane);
        v16h a0l = load_tile16(sAl + kc,               A_PITCH, lane);
        v16h a1l = load_tile16(sAl + 16 * A_PITCH + kc, A_PITCH, lane);
#pragma unroll
        for (int j = 0; j < 4; ++j) {
            v16h bh = load_tile16(Bh + (size_t)(16 * j) * DIMN + kc, DIMN, lane);
            v16h bl = load_tile16(Bl + (size_t)(16 * j) * DIMN + kc, DIMN, lane);
            acc[0][j] = WMMA_F16(a0h, bh, acc[0][j]);
            acc[0][j] = WMMA_F16(a0l, bh, acc[0][j]);
            acc[0][j] = WMMA_F16(a0h, bl, acc[0][j]);
            acc[1][j] = WMMA_F16(a1h, bh, acc[1][j]);
            acc[1][j] = WMMA_F16(a1l, bh, acc[1][j]);
            acc[1][j] = WMMA_F16(a1h, bl, acc[1][j]);
        }
    }

    // ---- epilogue: bias add, repack via wave-private LDS, b128 stores -----
    float* myOut = sOut + wid * (32 * SOUT_PITCH);
    const int col  = lane & 15;
    const int rsel = (lane < 16) ? 0 : 8;
#pragma unroll
    for (int j = 0; j < 4; ++j) {
        const float bj = bias[e * DIMN + c0 + 16 * j + col];
#pragma unroll
        for (int rt = 0; rt < 2; ++rt) {
#pragma unroll
            for (int i = 0; i < 8; ++i) {
                myOut[(16 * rt + rsel + i) * SOUT_PITCH + 16 * j + col] =
                    acc[rt][j][i] + bj;
            }
        }
    }
    // Read back row-major: each half-wave covers one 256B-contiguous row.
    float* orow = out + ((size_t)e * NTOK + m0) * DIMN + c0;
#pragma unroll
    for (int t = 0; t < 16; ++t) {
        int id  = t * 32 + lane;       // 512 float4-chunks: 32 rows x 16 chunks
        int row = id >> 4;
        int ch4 = (id & 15) << 2;
        float4 v = *(const float4*)(myOut + row * SOUT_PITCH + ch4);
        *(float4*)(orow + (size_t)row * DIMN + ch4) = v;
    }
}

// ---------------------------------------------------------------------------
extern "C" void kernel_launch(void* const* d_in, const int* in_sizes, int n_in,
                              void* d_out, int out_size, void* d_ws, size_t ws_size,
                              hipStream_t stream) {
    const float* x = (const float*)d_in[0];   // [NTOK, DIMN]
    const float* W = (const float*)d_in[1];   // [NEXP, DIMN, DIMN]
    const float* b = (const float*)d_in[2];   // [NEXP, DIMN]
    float* out = (float*)d_out;               // [NEXP, NTOK, DIMN]

    // Workspace layout (32 MiB): xh | xl | wh | wl  (f16 planes)
    char* ws = (char*)d_ws;
    const size_t xBytes = (size_t)NTOK * DIMN * sizeof(_Float16);
    const size_t wBytes = (size_t)NEXP * DIMN * DIMN * sizeof(_Float16);
    _Float16* xh = (_Float16*)(ws);
    _Float16* xl = (_Float16*)(ws + xBytes);
    _Float16* wh = (_Float16*)(ws + 2 * xBytes);
    _Float16* wl = (_Float16*)(ws + 2 * xBytes + wBytes);

    const int nx = NTOK * DIMN;
    const int nw = NEXP * DIMN * DIMN;
    const int thr = 256;
    split_f32_to_f16hl<<<(nx / 4 + thr - 1) / thr, thr, 0, stream>>>(x, xh, xl, nx);
    split_f32_to_f16hl<<<(nw / 4 + thr - 1) / thr, thr, 0, stream>>>(W, wh, wl, nw);

    // One block per (expert, 32-token tile): 16 * 256 = 4096 blocks, 8 waves.
    expert_gemm_wmma<<<NEXP * (NTOK / 32), 256, SMEM_BYTES, stream>>>(
        xh, xl, wh, wl, b, out);
}
